// NystromAttention_5291399708992
// MI455X (gfx1250) — compile-verified, matrix operand order checked by reading, not run on hardware
//
#include <hip/hip_runtime.h>

// ---------------------------------------------------------------------------
// Nystrom attention for MI455X (gfx1250, wave32)
// WMMA f32_16x16x32_f16 everywhere + async global->LDS staging (ASYNCcnt)
// ---------------------------------------------------------------------------

typedef _Float16 f16;
typedef __attribute__((ext_vector_type(16))) _Float16 v16h;
typedef __attribute__((ext_vector_type(8)))  _Float16 v8h;
typedef __attribute__((ext_vector_type(8)))  float    v8f;

#define DEVINL __device__ __forceinline__

#define BB     4
#define NH     8
#define NN     4096
#define DH     64
#define MM     256
#define LLM    16        // tokens per landmark
#define DIMD   512
#define NQKV   1536
#define SCALEQ 0.125f
#define KSZ    33

// ---- WMMA helpers ----------------------------------------------------------

DEVINL v8f wmma16(v16h a, v16h b, v8f c) {
  return __builtin_amdgcn_wmma_f32_16x16x32_f16(
      /*neg_a=*/false, a, /*neg_b=*/false, b,
      /*c_mod=*/(short)0, c, /*reuse_a=*/false, /*reuse_b=*/false);
}

// Load a 16x32 f16 fragment from a row-major [ld] matrix at (row0, k0).
// ISA layout: lanes 0-15 hold K=[0..7]+[16..23]; lanes 16-31 K=[8..15]+[24..31]
// -> two contiguous 16B loads per lane (global_load_b128 or ds_load_b128).
DEVINL v16h frag_ld(const f16* __restrict__ p, int ld, int row0, int k0, int lane) {
  int r  = row0 + (lane & 15);
  int kb = k0 + ((lane >> 4) << 3);
  const f16* base = p + (long)r * ld + kb;
  v8h lo = *(const v8h*)(base);
  v8h hi = *(const v8h*)(base + 16);
  v16h o;
#pragma unroll
  for (int i = 0; i < 8; ++i) { o[i] = lo[i]; o[i + 8] = hi[i]; }
  return o;
}

// ---- CDNA5 async global->LDS (tracked by ASYNCcnt) -------------------------
// Per-lane 16B transfer: LDS byte address in a VGPR, 64-bit global address in
// a VGPR pair. Generic pointers to LDS carry the LDS offset in their low 32b.

DEVINL void async_g2l_b128(const f16* g, void* l) {
  unsigned loff = (unsigned)(unsigned long long)l;
  asm volatile("global_load_async_to_lds_b128 %0, %1, off"
               :: "v"(loff), "v"(g)
               : "memory");
}
DEVINL void async_wait0() {
  asm volatile("s_wait_asynccnt 0x0" ::: "memory");
}

// reductions over the 16 lanes of a wave32 half (C-fragment row group)
DEVINL float hmax16(float v) {
  v = fmaxf(v, __shfl_xor(v, 1, 32));
  v = fmaxf(v, __shfl_xor(v, 2, 32));
  v = fmaxf(v, __shfl_xor(v, 4, 32));
  v = fmaxf(v, __shfl_xor(v, 8, 32));
  return v;
}
DEVINL float hsum16(float v) {
  v += __shfl_xor(v, 1, 32);
  v += __shfl_xor(v, 2, 32);
  v += __shfl_xor(v, 4, 32);
  v += __shfl_xor(v, 8, 32);
  return v;
}

// ---- conversion kernels ----------------------------------------------------

__global__ void k_f32_to_f16(const float* __restrict__ in, f16* __restrict__ out, int n) {
  int i = blockIdx.x * blockDim.x + threadIdx.x;
  if (i < n) out[i] = (f16)in[i];
}

__global__ void k_transpose_f16(const float* __restrict__ in, f16* __restrict__ out,
                                int rows, int cols) {
  int i = blockIdx.x * blockDim.x + threadIdx.x;
  if (i < rows * cols) {
    int r = i / cols, c = i - r * cols;
    out[(long)c * rows + r] = (f16)in[i];
  }
}

// ---- K1: qkv GEMM (x @ w_qkv), scatter into head-major q/k/v (+vT) ---------
// 128x128 tile per block; B (w_qkv^T) slice staged to LDS via async DMA,
// double-buffered; A fragments software-pipelined in registers.

__global__ void k_qkv_gemm(const f16* __restrict__ X, const f16* __restrict__ Wt,
                           f16* __restrict__ q16, f16* __restrict__ k16,
                           f16* __restrict__ v16v, f16* __restrict__ vT16) {
  __shared__ __align__(16) f16 Bt[2][128][32];   // 16 KB
  int wave = threadIdx.x >> 5, lane = threadIdx.x & 31;
  int tid = threadIdx.x;
  int row0 = blockIdx.x * 128 + wave * 16;
  int col0 = blockIdx.y * 128;
  const f16* Bb = Wt + (long)col0 * DIMD;

  auto stageB = [&](int buf, int k0) {
#pragma unroll
    for (int j = 0; j < 2; ++j) {             // 128*32/8 = 512 xfers, 2/thread
      int t = tid + j * 256;
      int r = t >> 2, co = (t & 3) * 8;
      async_g2l_b128(Bb + (long)r * DIMD + k0 + co, &Bt[buf][r][co]);
    }
  };

  v8f acc[8] = {};
  stageB(0, 0);
  v16h a_cur = frag_ld(X, DIMD, row0, 0, lane);
  int buf = 0;
  for (int k = 0; k < DIMD; k += 32) {
    async_wait0();
    __syncthreads();
    if (k + 32 < DIMD) stageB(buf ^ 1, k + 32);
    v16h a_nxt = a_cur;
    if (k + 32 < DIMD) a_nxt = frag_ld(X, DIMD, row0, k + 32, lane);
#pragma unroll
    for (int t = 0; t < 8; ++t)
      acc[t] = wmma16(a_cur, frag_ld(&Bt[buf][0][0], 32, t * 16, 0, lane), acc[t]);
    a_cur = a_nxt;
    __syncthreads();
    buf ^= 1;
  }

  int half = lane >> 4, lc = lane & 15;
#pragma unroll
  for (int t = 0; t < 8; ++t) {
    int c = col0 + t * 16 + lc;
    int sect = c >> 9;        // 0:q 1:k 2:v
    int inner = c & 511;
    int h = inner >> 6, d = inner & 63;
#pragma unroll
    for (int v = 0; v < 8; ++v) {
      int row = row0 + v + 8 * half;
      int b = row >> 12, i = row & 4095;
      float val = acc[t][v];
      long base = (((long)(b * NH + h) * NN) + i) * DH + d;
      if (sect == 0)      q16[base] = (f16)(val * SCALEQ);
      else if (sect == 1) k16[base] = (f16)val;
      else {
        v16v[base] = (f16)val;
        vT16[((long)(b * NH + h) * DH + d) * NN + i] = (f16)val;
      }
    }
  }
}

// ---- K2: landmark means ----------------------------------------------------

__global__ void k_landmarks(const f16* __restrict__ q16, const f16* __restrict__ k16,
                            f16* __restrict__ qland, f16* __restrict__ kland) {
  int idx = blockIdx.x * blockDim.x + threadIdx.x;
  const int half_total = BB * NH * MM * DH;
  int which = idx >= half_total;
  int r = which ? idx - half_total : idx;
  int d = r & 63, m = (r >> 6) & 255, bh = r >> 14;
  const f16* src = which ? k16 : q16;
  float s = 0.f;
#pragma unroll
  for (int t = 0; t < LLM; ++t)
    s += (float)src[((long)bh * NN + m * LLM + t) * DH + d];
  (which ? kland : qland)[((long)bh * MM + m) * DH + d] = (f16)(s * (1.0f / LLM));
}

// ---- K3: residual depthwise conv along seq ---------------------------------

__global__ void k_dwconv(const f16* __restrict__ v16v, const float* __restrict__ cw,
                         float* __restrict__ res) {
  int idx = blockIdx.x * blockDim.x + threadIdx.x;   // (bh, i, d)
  int d = idx & 63, i = (idx >> 6) & 4095, bh = idx >> 18;
  int h = bh & 7;
  float s = 0.f;
#pragma unroll 1
  for (int t = 0; t < KSZ; ++t) {
    int j = i + t - KSZ / 2;
    if (j >= 0 && j < NN)
      s += cw[h * KSZ + t] * (float)v16v[((long)bh * NN + j) * DH + d];
  }
  res[idx] = s;
}

// ---- K4a: attn2 = softmax(q_land @ k_land^T) -------------------------------

__global__ void k_attn2(const f16* __restrict__ qland, const f16* __restrict__ kland,
                        float* __restrict__ attn2, f16* __restrict__ X16) {
  int bh = blockIdx.x >> 1, rb = blockIdx.x & 1;
  int wave = threadIdx.x >> 5, lane = threadIdx.x & 31;
  int row0 = rb * 128 + wave * 16;
  const f16* qa = qland + (long)bh * MM * DH;
  const f16* ka = kland + (long)bh * MM * DH;
  v8f acc[16] = {};
  for (int k = 0; k < DH; k += 32) {
    v16h a = frag_ld(qa, DH, row0, k, lane);
#pragma unroll
    for (int t = 0; t < 16; ++t)
      acc[t] = wmma16(a, frag_ld(ka, DH, t * 16, k, lane), acc[t]);
  }
  int half = lane >> 4, lc = lane & 15;
#pragma unroll
  for (int v = 0; v < 8; ++v) {
    float m = -1e30f;
#pragma unroll
    for (int t = 0; t < 16; ++t) m = fmaxf(m, acc[t][v]);
    m = hmax16(m);
    float s = 0.f;
#pragma unroll
    for (int t = 0; t < 16; ++t) { float e = __expf(acc[t][v] - m); acc[t][v] = e; s += e; }
    s = hsum16(s);
    float inv = 1.0f / s;
    int row = row0 + v + 8 * half;
#pragma unroll
    for (int t = 0; t < 16; ++t) {
      float p = acc[t][v] * inv;
      long o = (long)bh * MM * MM + (long)row * MM + (t * 16 + lc);
      attn2[o] = p;
      X16[o] = (f16)p;
    }
  }
}

// ---- K4b: pinv init (global max row/col abs-sums, z = x^T / s) -------------

__global__ void k_pinv_zero(float* g) { if (threadIdx.x < 2) g[threadIdx.x] = 0.f; }

__global__ void k_pinv_maxsums(const float* __restrict__ attn2, float* __restrict__ gmax) {
  int bh = blockIdx.x, i = threadIdx.x;
  const float* a = attn2 + (long)bh * MM * MM;
  float rs = 0.f, cs = 0.f;
  for (int j = 0; j < MM; ++j) {
    rs += fabsf(a[i * MM + j]);
    cs += fabsf(a[j * MM + i]);
  }
  __shared__ float sm[256];
  sm[i] = rs; __syncthreads();
  for (int s = 128; s > 0; s >>= 1) { if (i < s) sm[i] = fmaxf(sm[i], sm[i + s]); __syncthreads(); }
  if (i == 0) atomicMax((unsigned int*)&gmax[0], __float_as_uint(sm[0]));
  __syncthreads();
  sm[i] = cs; __syncthreads();
  for (int s = 128; s > 0; s >>= 1) { if (i < s) sm[i] = fmaxf(sm[i], sm[i + s]); __syncthreads(); }
  if (i == 0) atomicMax((unsigned int*)&gmax[1], __float_as_uint(sm[0]));
}

__global__ void k_pinv_init(const float* __restrict__ attn2, const float* __restrict__ gmax,
                            f16* __restrict__ Z, f16* __restrict__ Zt) {
  int bh = blockIdx.x, i = threadIdx.x;
  float scale = 1.0f / (gmax[0] * gmax[1]);
  const float* a = attn2 + (long)bh * MM * MM;
  long base = (long)bh * MM * MM + (long)i * MM;
  for (int j = 0; j < MM; ++j) {
    Z[base + j]  = (f16)(a[j * MM + i] * scale);   // Z = x^T / s (A-form)
    Zt[base + j] = (f16)(a[i * MM + j] * scale);   // Z^T        (B-storage)
  }
}

// ---- K4c: batched GEMM  C = A @ B^T  with dual (a*acc + b*I) epilogue ------
// B (N x K row-major) staged to LDS via async DMA, double-buffered.

template <int NT>
__global__ void k_gemm_bt(const f16* __restrict__ A, const f16* __restrict__ Bm,
                          f16* __restrict__ O1, f16* __restrict__ O2,
                          int K, int lda, int ldb,
                          long sA, long sB, long sO1, int ldo1, long sO2, int ldo2,
                          float a1, float b1, float a2, float b2, int rowBlocks) {
  __shared__ __align__(16) f16 Bt[2][NT * 16][32];
  constexpr int PER = (NT * 64) / 256;   // b128 transfers per thread per stage
  static_assert(NT % 4 == 0, "NT must be a multiple of 4");

  int batch = blockIdx.x / rowBlocks, rb = blockIdx.x % rowBlocks;
  int wave = threadIdx.x >> 5, lane = threadIdx.x & 31;
  int tid = threadIdx.x;
  int row0 = rb * 128 + wave * 16;
  const f16* Ab = A + batch * sA;
  const f16* Bb = Bm + batch * sB;

  auto stageB = [&](int buf, int k0) {
#pragma unroll
    for (int j = 0; j < PER; ++j) {
      int t = tid + j * 256;
      int r = t >> 2, co = (t & 3) * 8;
      async_g2l_b128(Bb + (long)r * ldb + k0 + co, &Bt[buf][r][co]);
    }
  };

  v8f acc[NT] = {};
  stageB(0, 0);
  v16h a_cur = frag_ld(Ab, lda, row0, 0, lane);
  int buf = 0;
  for (int k = 0; k < K; k += 32) {
    async_wait0();
    __syncthreads();
    if (k + 32 < K) stageB(buf ^ 1, k + 32);
    v16h a_nxt = a_cur;
    if (k + 32 < K) a_nxt = frag_ld(Ab, lda, row0, k + 32, lane);
#pragma unroll
    for (int t = 0; t < NT; ++t)
      acc[t] = wmma16(a_cur, frag_ld(&Bt[buf][0][0], 32, t * 16, 0, lane), acc[t]);
    a_cur = a_nxt;
    __syncthreads();
    buf ^= 1;
  }

  int half = lane >> 4, lc = lane & 15;
#pragma unroll
  for (int t = 0; t < NT; ++t) {
    int col = t * 16 + lc;
#pragma unroll
    for (int v = 0; v < 8; ++v) {
      int row = row0 + v + 8 * half;
      float val = acc[t][v];
      float dia = (row == col) ? 1.0f : 0.0f;
      if (O1) O1[batch * sO1 + (long)row * ldo1 + col] = (f16)(a1 * val + b1 * dia);
      if (O2) O2[batch * sO2 + (long)col * ldo2 + row] = (f16)(a2 * val + b2 * dia);
    }
  }
}

// ---- K5: flash-style  avT = (softmax(q_land @ k^T) @ v)^T ------------------
// One block per (b,h); K and V^T chunks staged via async DMA, double-buffered;
// q_land fragments hoisted (loop-invariant).

__global__ void k_flash_attn3v(const f16* __restrict__ qland, const f16* __restrict__ k16,
                               const f16* __restrict__ vT16, f16* __restrict__ avT) {
  __shared__ __align__(16) f16 Kt[2][64][64];   // 16 KB
  __shared__ __align__(16) f16 Vt[2][64][64];   // 16 KB
  __shared__ __align__(16) f16 Pl[8][32][64];   // 32 KB
  int bh = blockIdx.x;
  int wave = threadIdx.x >> 5, lane = threadIdx.x & 31;
  int tid = threadIdx.x;
  int half = lane >> 4, lc = lane & 15;
  int r0 = wave * 32;
  const f16* qa = qland + (long)bh * MM * DH;
  const f16* ka = k16 + (long)bh * NN * DH;
  const f16* va = vT16 + (long)bh * DH * NN;

  auto stage = [&](int buf, int c) {
#pragma unroll
    for (int j = 0; j < 2; ++j) {             // K rows: 64x64 halves
      int t = tid + j * 256;
      int r = t >> 3, co = (t & 7) * 8;
      async_g2l_b128(ka + (long)(c + r) * DH + co, &Kt[buf][r][co]);
    }
#pragma unroll
    for (int j = 0; j < 2; ++j) {             // V^T rows (d): 64x64 halves
      int t = tid + j * 256;
      int r = t >> 3, co = (t & 7) * 8;
      async_g2l_b128(va + (long)r * NN + c + co, &Vt[buf][r][co]);
    }
  };

  // hoisted q_land fragments (invariant across chunks)
  v16h aq0[2], aq1[2];
#pragma unroll
  for (int ks = 0; ks < 2; ++ks) {
    aq0[ks] = frag_ld(qa, DH, r0, ks * 32, lane);
    aq1[ks] = frag_ld(qa, DH, r0 + 16, ks * 32, lane);
  }

  v8f accO[2][4] = {};
  float mst[2][8], lst[2][8];
#pragma unroll
  for (int rt = 0; rt < 2; ++rt)
#pragma unroll
    for (int v = 0; v < 8; ++v) { mst[rt][v] = -1e30f; lst[rt][v] = 0.f; }

  stage(0, 0);
  int buf = 0;
  for (int c = 0; c < NN; c += 64) {
    async_wait0();
    __syncthreads();
    if (c + 64 < NN) stage(buf ^ 1, c + 64);

    v8f S[2][4] = {};
#pragma unroll
    for (int ks = 0; ks < 2; ++ks) {
#pragma unroll
      for (int ct = 0; ct < 4; ++ct) {
        v16h bf = frag_ld(&Kt[buf][0][0], 64, ct * 16, ks * 32, lane);
        S[0][ct] = wmma16(aq0[ks], bf, S[0][ct]);
        S[1][ct] = wmma16(aq1[ks], bf, S[1][ct]);
      }
    }
    // online softmax update + stage P in LDS
#pragma unroll
    for (int rt = 0; rt < 2; ++rt) {
#pragma unroll
      for (int v = 0; v < 8; ++v) {
        float rm = -1e30f;
#pragma unroll
        for (int ct = 0; ct < 4; ++ct) rm = fmaxf(rm, S[rt][ct][v]);
        rm = hmax16(rm);
        float mn = fmaxf(mst[rt][v], rm);
        float alpha = __expf(mst[rt][v] - mn);
        mst[rt][v] = mn;
        float rs = 0.f;
#pragma unroll
        for (int ct = 0; ct < 4; ++ct) { float e = __expf(S[rt][ct][v] - mn); S[rt][ct][v] = e; rs += e; }
        rs = hsum16(rs);
        lst[rt][v] = lst[rt][v] * alpha + rs;
#pragma unroll
        for (int dt = 0; dt < 4; ++dt) accO[rt][dt][v] = accO[rt][dt][v] * alpha;
        int lrow = rt * 16 + v + 8 * half;
#pragma unroll
        for (int ct = 0; ct < 4; ++ct) Pl[wave][lrow][ct * 16 + lc] = (f16)S[rt][ct][v];
      }
    }
    // accO += P @ v_chunk  (B from staged V^T)
#pragma unroll
    for (int ks = 0; ks < 2; ++ks) {
      v16h a0 = frag_ld(&Pl[wave][0][0], 64, 0, ks * 32, lane);
      v16h a1 = frag_ld(&Pl[wave][0][0], 64, 16, ks * 32, lane);
#pragma unroll
      for (int dt = 0; dt < 4; ++dt) {
        v16h bf = frag_ld(&Vt[buf][0][0], 64, dt * 16, ks * 32, lane);
        accO[0][dt] = wmma16(a0, bf, accO[0][dt]);
        accO[1][dt] = wmma16(a1, bf, accO[1][dt]);
      }
    }
    __syncthreads();
    buf ^= 1;
  }
  // write avT[bh][d][m] = acc/l
#pragma unroll
  for (int rt = 0; rt < 2; ++rt)
#pragma unroll
    for (int v = 0; v < 8; ++v) {
      float inv = 1.0f / lst[rt][v];
      int grow = r0 + rt * 16 + v + 8 * half;
#pragma unroll
      for (int dt = 0; dt < 4; ++dt) {
        int d = dt * 16 + lc;
        avT[(long)bh * DH * MM + (long)d * MM + grow] = (f16)(accO[rt][dt][v] * inv);
      }
    }
}

// ---- K6: out_heads = softmax(q @ k_land^T) @ W2 + conv residual ------------
// k_land staged to LDS via async DMA (shared by all 8 waves).

__global__ void k_attn1_out(const f16* __restrict__ q16, const f16* __restrict__ kland,
                            const f16* __restrict__ W2t, const float* __restrict__ res,
                            f16* __restrict__ outh) {
  __shared__ __align__(16) f16 Klt[256][64];    // 32 KB
  __shared__ __align__(16) f16 Pl[8][16][256];  // 64 KB
  int bidx = blockIdx.x;
  int nb = bidx & 31, bh = bidx >> 5;
  int wave = threadIdx.x >> 5, lane = threadIdx.x & 31;
  int tid = threadIdx.x;
  int half = lane >> 4, lc = lane & 15;
  int row0 = nb * 128 + wave * 16;
  const f16* qa = q16 + (long)bh * NN * DH;
  const f16* ka = kland + (long)bh * MM * DH;
  const f16* wb = W2t + (long)bh * DH * MM;

  // stage k_land (256x64)
#pragma unroll
  for (int j = 0; j < 8; ++j) {                 // 2048 xfers, 8/thread
    int t = tid + j * 256;
    int r = t >> 3, co = (t & 7) * 8;
    async_g2l_b128(ka + (long)r * DH + co, &Klt[r][co]);
  }
  async_wait0();
  __syncthreads();

  v8f S[16] = {};
  for (int k = 0; k < DH; k += 32) {
    v16h a = frag_ld(qa, DH, row0, k, lane);
#pragma unroll
    for (int ct = 0; ct < 16; ++ct)
      S[ct] = wmma16(a, frag_ld(&Klt[0][0], 64, ct * 16, k, lane), S[ct]);
  }
#pragma unroll
  for (int v = 0; v < 8; ++v) {
    float m = -1e30f;
#pragma unroll
    for (int ct = 0; ct < 16; ++ct) m = fmaxf(m, S[ct][v]);
    m = hmax16(m);
    float s = 0.f;
#pragma unroll
    for (int ct = 0; ct < 16; ++ct) { float e = __expf(S[ct][v] - m); S[ct][v] = e; s += e; }
    s = hsum16(s);
    float inv = 1.0f / s;
    int lrow = v + 8 * half;
#pragma unroll
    for (int ct = 0; ct < 16; ++ct) Pl[wave][lrow][ct * 16 + lc] = (f16)(S[ct][v] * inv);
  }
  // out(16x64) = P(16x256) @ W2(256x64)
  v8f acc[4] = {};
  for (int k = 0; k < MM; k += 32) {
    v16h a = frag_ld(&Pl[wave][0][0], 256, 0, k, lane);
#pragma unroll
    for (int dt = 0; dt < 4; ++dt)
      acc[dt] = wmma16(a, frag_ld(wb, MM, dt * 16, k, lane), acc[dt]);
  }
  int b = bh >> 3, h = bh & 7;
#pragma unroll
  for (int dt = 0; dt < 4; ++dt) {
    int d = dt * 16 + lc;
#pragma unroll
    for (int v = 0; v < 8; ++v) {
      int i = row0 + v + 8 * half;
      float val = acc[dt][v] + res[((long)bh * NN + i) * DH + d];
      outh[((long)b * NN + i) * DIMD + h * DH + d] = (f16)val;
    }
  }
}

// ---- K7: final projection  out = out_heads @ w_out (f32 output) ------------

__global__ void k_out_gemm(const f16* __restrict__ Xh, const f16* __restrict__ Wt,
                           float* __restrict__ out) {
  __shared__ __align__(16) f16 Bt[2][128][32];   // 16 KB
  int wave = threadIdx.x >> 5, lane = threadIdx.x & 31;
  int tid = threadIdx.x;
  int row0 = blockIdx.x * 128 + wave * 16;
  int col0 = blockIdx.y * 128;
  const f16* Bb = Wt + (long)col0 * DIMD;

  auto stageB = [&](int buf, int k0) {
#pragma unroll
    for (int j = 0; j < 2; ++j) {
      int t = tid + j * 256;
      int r = t >> 2, co = (t & 3) * 8;
      async_g2l_b128(Bb + (long)r * DIMD + k0 + co, &Bt[buf][r][co]);
    }
  };

  v8f acc[8] = {};
  stageB(0, 0);
  v16h a_cur = frag_ld(Xh, DIMD, row0, 0, lane);
  int buf = 0;
  for (int k = 0; k < DIMD; k += 32) {
    async_wait0();
    __syncthreads();
    if (k + 32 < DIMD) stageB(buf ^ 1, k + 32);
    v16h a_nxt = a_cur;
    if (k + 32 < DIMD) a_nxt = frag_ld(Xh, DIMD, row0, k + 32, lane);
#pragma unroll
    for (int t = 0; t < 8; ++t)
      acc[t] = wmma16(a_cur, frag_ld(&Bt[buf][0][0], 32, t * 16, 0, lane), acc[t]);
    a_cur = a_nxt;
    __syncthreads();
    buf ^= 1;
  }

  int half = lane >> 4, lc = lane & 15;
#pragma unroll
  for (int t = 0; t < 8; ++t)
#pragma unroll
    for (int v = 0; v < 8; ++v)
      out[(long)(row0 + v + 8 * half) * DIMD + (col0 + t * 16 + lc)] = acc[t][v];
}

// ---------------------------------------------------------------------------

extern "C" void kernel_launch(void* const* d_in, const int* in_sizes, int n_in,
                              void* d_out, int out_size, void* d_ws, size_t ws_size,
                              hipStream_t stream) {
  const float* x     = (const float*)d_in[0];
  const float* wqkv  = (const float*)d_in[1];
  const float* wout  = (const float*)d_in[2];
  const float* convw = (const float*)d_in[3];
  float* out = (float*)d_out;

  char* wsb = (char*)d_ws;
  size_t off = 0;
  auto alloc = [&](size_t bytes) -> void* {
    void* p = wsb + off;
    off = (off + bytes + 255) & ~(size_t)255;
    return p;
  };

  const size_t headsz = (size_t)BB * NH * NN * DH;      // 8.4M elems
  const size_t mm2    = (size_t)BB * NH * MM * MM;      // 2.1M elems

  f16* x16    = (f16*)alloc((size_t)BB * NN * DIMD * 2);
  f16* wqkvT  = (f16*)alloc((size_t)NQKV * DIMD * 2);
  f16* woutT  = (f16*)alloc((size_t)DIMD * DIMD * 2);
  f16* q16    = (f16*)alloc(headsz * 2);
  f16* k16    = (f16*)alloc(headsz * 2);
  f16* v16v   = (f16*)alloc(headsz * 2);
  f16* vT16   = (f16*)alloc(headsz * 2);
  f16* qland  = (f16*)alloc((size_t)BB * NH * MM * DH * 2);
  f16* kland  = (f16*)alloc((size_t)BB * NH * MM * DH * 2);
  float* attn2 = (float*)alloc(mm2 * 4);
  f16* X16    = (f16*)alloc(mm2 * 2);
  f16* Zb0    = (f16*)alloc(mm2 * 2);
  f16* Zb1    = (f16*)alloc(mm2 * 2);
  f16* Zt0    = (f16*)alloc(mm2 * 2);
  f16* Zt1    = (f16*)alloc(mm2 * 2);
  f16* XZ     = (f16*)alloc(mm2 * 2);
  f16* Tt0    = (f16*)alloc(mm2 * 2);
  f16* Tt1    = (f16*)alloc(mm2 * 2);
  float* gmax = (float*)alloc(256);
  f16* avT    = (f16*)alloc((size_t)BB * NH * DH * MM * 2);
  f16* W2t    = (f16*)alloc((size_t)BB * NH * DH * MM * 2);
  float* res  = (float*)alloc(headsz * 4);
  f16* outh   = (f16*)alloc((size_t)BB * NN * DIMD * 2);

  f16* Zb[2] = {Zb0, Zb1};
  f16* Zt[2] = {Zt0, Zt1};

  const long sMM = (long)MM * MM;          // 65536
  const long sDM = (long)DH * MM;          // 16384

  // 0) fp32 -> f16 staging
  int nx = BB * NN * DIMD;
  k_f32_to_f16<<<(nx + 255) / 256, 256, 0, stream>>>(x, x16, nx);
  k_transpose_f16<<<(DIMD * NQKV + 255) / 256, 256, 0, stream>>>(wqkv, wqkvT, DIMD, NQKV);
  k_transpose_f16<<<(DIMD * DIMD + 255) / 256, 256, 0, stream>>>(wout, woutT, DIMD, DIMD);

  // 1) qkv projection (WMMA + async LDS staging)
  k_qkv_gemm<<<dim3(128, 12), 256, 0, stream>>>(x16, wqkvT, q16, k16, v16v, vT16);

  // 2) landmarks, 3) conv residual
  k_landmarks<<<(2 * BB * NH * MM * DH) / 256, 256, 0, stream>>>(q16, k16, qland, kland);
  k_dwconv<<<(int)(headsz / 256), 256, 0, stream>>>(v16v, convw, res);

  // 4) attn2 + Moore-Penrose iterations (all WMMA)
  k_attn2<<<64, 256, 0, stream>>>(qland, kland, attn2, X16);
  k_pinv_zero<<<1, 32, 0, stream>>>(gmax);
  k_pinv_maxsums<<<32, 256, 0, stream>>>(attn2, gmax);
  k_pinv_init<<<32, 256, 0, stream>>>(attn2, gmax, Zb[0], Zt[0]);

  for (int it = 0; it < 6; ++it) {
    int cur = it & 1, nxt = cur ^ 1;
    // XZ = X@Z (row-major); T1^T = (7I - XZ)^T
    k_gemm_bt<16><<<64, 256, 0, stream>>>(X16, Zt[cur], XZ, Tt0, MM, MM, MM,
        sMM, sMM, sMM, MM, sMM, MM, 1.f, 0.f, -1.f, 7.f, 2);
    // T2^T = (15I - XZ@T1)^T
    k_gemm_bt<16><<<64, 256, 0, stream>>>(XZ, Tt0, (f16*)nullptr, Tt1, MM, MM, MM,
        sMM, sMM, 0, 0, sMM, MM, 0.f, 0.f, -1.f, 15.f, 2);
    // T3^T = (13I - XZ@T2)^T
    k_gemm_bt<16><<<64, 256, 0, stream>>>(XZ, Tt1, (f16*)nullptr, Tt0, MM, MM, MM,
        sMM, sMM, 0, 0, sMM, MM, 0.f, 0.f, -1.f, 13.f, 2);
    // Z_next = 0.25 * Z @ T3  (both row-major and transposed forms)
    k_gemm_bt<16><<<64, 256, 0, stream>>>(Zb[cur], Tt0, Zb[nxt], Zt[nxt], MM, MM, MM,
        sMM, sMM, sMM, MM, sMM, MM, 0.25f, 0.f, 0.25f, 0.f, 2);
  }

  // 5) av = softmax(q_land @ k^T) @ v  (flash-style, async-staged K/V^T)
  k_flash_attn3v<<<32, 256, 0, stream>>>(qland, k16, vT16, avT);

  // 6) W2^T = (Z @ av)^T
  k_gemm_bt<4><<<64, 256, 0, stream>>>(Zb[0], avT, (f16*)nullptr, W2t, MM, MM, MM,
      sMM, sDM, 0, 0, sDM, MM, 0.f, 0.f, 1.f, 0.f, 2);

  // 7) out_heads = softmax(q @ k_land^T) @ W2 + conv residual
  k_attn1_out<<<BB * NH * (NN / 128), 256, 0, stream>>>(q16, kland, W2t, res, outh);

  // 8) final projection -> fp32 output
  k_out_gemm<<<dim3(128, 4), 256, 0, stream>>>(outh, woutT, out);

  (void)in_sizes; (void)n_in; (void)out_size; (void)ws_size;
}
